// MultiheadGQA_10230612099680
// MI455X (gfx1250) — matrix operand order repaired and checked by reading, not compiled, not run on hardware
//
#include <hip/hip_runtime.h>

// ---------------------------------------------------------------------------
// MI455X (gfx1250) attention block: bf16 WMMA GEMMs + flash attention + LN.
// Matmuls lower to v_wmma_f32_16x16x32_bf16 (wave32). K-tiles are double-
// buffered; A/Q/K tiles are staged with the CDNA5 async global->LDS DMA path
// (global_load_async_to_lds_b128, ASYNCcnt + s_wait_asynccnt).
// ---------------------------------------------------------------------------

typedef __attribute__((ext_vector_type(16))) __bf16        v16bf;
typedef __attribute__((ext_vector_type(8)))  float         v8f;
typedef __attribute__((ext_vector_type(4)))  unsigned int  v4u;
typedef __attribute__((ext_vector_type(8)))  unsigned int  v8u;
typedef unsigned int   u32;
typedef unsigned short u16;
typedef int v4i_g __attribute__((vector_size(16)));   // matches builtin proto

#if __has_builtin(__builtin_amdgcn_global_load_async_to_lds_b128) && \
    __has_builtin(__builtin_amdgcn_s_wait_asynccnt)
#define HAVE_ASYNC 1
#else
#define HAVE_ASYNC 0
#endif

#if HAVE_ASYNC
__device__ __forceinline__ void async_cp16(const void* g, u32* l) {
  __builtin_amdgcn_global_load_async_to_lds_b128(
      (__attribute__((address_space(1))) v4i_g*)g,
      (__attribute__((address_space(3))) v4i_g*)l, 0, 0);
}
__device__ __forceinline__ void async_wait0() {
  __builtin_amdgcn_s_wait_asynccnt(0);
}
#else
__device__ __forceinline__ void async_wait0() {}
#endif

__device__ __forceinline__ u16 f2bf(float f) {
  u32 u = __float_as_uint(f);
  u32 r = (u + 0x7FFFu + ((u >> 16) & 1u)) >> 16;   // round-to-nearest-even
  return (u16)r;
}

__device__ __forceinline__ v8f wmma_bf16(v8u a, v8u b, v8f c) {
  return __builtin_amdgcn_wmma_f32_16x16x32_bf16(
      false, __builtin_bit_cast(v16bf, a),
      false, __builtin_bit_cast(v16bf, b),
      (short)0, c, false, false);
}

__device__ __forceinline__ float hmax16(float v) {
#pragma unroll
  for (int o = 1; o < 16; o <<= 1) v = fmaxf(v, __shfl_xor(v, o, 32));
  return v;
}
__device__ __forceinline__ float hsum16(float v) {
#pragma unroll
  for (int o = 1; o < 16; o <<= 1) v += __shfl_xor(v, o, 32);
  return v;
}
__device__ __forceinline__ float wsum32(float v) {
#pragma unroll
  for (int o = 16; o > 0; o >>= 1) v += __shfl_xor(v, o, 32);
  return v;
}

// ---------------------------------------------------------------------------
// fp32 -> bf16 conversion (vectorized 4/thread)
// ---------------------------------------------------------------------------
__global__ void k_f32_to_bf16(const float* __restrict__ src,
                              u16* __restrict__ dst, int n4) {
  int i = blockIdx.x * blockDim.x + threadIdx.x;
  int stride = gridDim.x * blockDim.x;
  for (; i < n4; i += stride) {
    float4 f = reinterpret_cast<const float4*>(src)[i];
    uint2 o;
    o.x = (u32)f2bf(f.x) | ((u32)f2bf(f.y) << 16);
    o.y = (u32)f2bf(f.z) | ((u32)f2bf(f.w) << 16);
    reinterpret_cast<uint2*>(dst)[i] = o;
  }
}

// ---------------------------------------------------------------------------
// Generic bf16 GEMM:  C[M,N] = A[M,K] (row-major, lda) @ B[K,N] (row-major, ldb)
//                     + bias[N].  f32 and/or bf16 output.
// 128x128 block tile, BK=32, 8 waves, wave tile 32x64 (2x4 WMMA tiles).
// Double-buffered LDS; A tile staged via async global->LDS DMA,
// B tile double-buffered through registers (pair-interleave transform).
// ---------------------------------------------------------------------------
#define GBM 128
#define GBN 128
#define GBK 32
#define SA_STRIDE 20   // 16 K-pairs + pad (uints)
#define SB_STRIDE 132  // 128 cols + pad (uints)

__global__ __launch_bounds__(256)
void k_gemm_bf16(const u16* __restrict__ A, int lda,
                 const u16* __restrict__ B, int ldb,
                 const float* __restrict__ bias,
                 float* __restrict__ outF, u16* __restrict__ outH,
                 int M, int N, int K) {
  __shared__ u32 sA[2][GBM * SA_STRIDE];   // K-pair-packed rows of A
  __shared__ u32 sB[2][16 * SB_STRIDE];    // pair-interleaved B

  const int tid  = threadIdx.x;
  const int lane = tid & 31;
  const int wave = tid >> 5;           // 0..7
  const int wr   = wave >> 1;          // 0..3 : 32-row group
  const int wc   = wave & 1;           // 0..1 : 64-col group
  const int half = lane >> 4;          // 0/1
  const int l16  = lane & 15;
  const int bm   = blockIdx.y * GBM;
  const int bn   = blockIdx.x * GBN;

  const int a_row = tid >> 1;          // 0..127
  const int a_pg  = (tid & 1) * 2;     // v4u group 0/2 (pair base 0/8)
  const int b_p0  = (tid * 2) >> 5;    // unit 0 pair idx
  const int b_c0  = ((tid * 2) & 31) * 4;
  const int b_p1  = (tid * 2 + 1) >> 5;
  const int b_c1  = ((tid * 2 + 1) & 31) * 4;

  v8f acc[2][4];
#pragma unroll
  for (int mt = 0; mt < 2; ++mt)
#pragma unroll
    for (int nt = 0; nt < 4; ++nt) acc[mt][nt] = (v8f){0, 0, 0, 0, 0, 0, 0, 0};

  // ---- staging helpers ----
  auto loadB_unit = [&](int kk, int p, int col) -> v4u {
    const u16* g0 = B + (size_t)(kk + 2 * p) * ldb + bn + col;
    uint2 r0 = *reinterpret_cast<const uint2*>(g0);
    uint2 r1 = *reinterpret_cast<const uint2*>(g0 + ldb);
    v4u w;
    w.x = (r0.x & 0xFFFFu) | (r1.x << 16);
    w.y = (r0.x >> 16)     | (r1.x & 0xFFFF0000u);
    w.z = (r0.y & 0xFFFFu) | (r1.y << 16);
    w.w = (r0.y >> 16)     | (r1.y & 0xFFFF0000u);
    return w;
  };

  const int nk = K / GBK;
  v4u bR0, bR1;        // B tile register double buffer
#if !HAVE_ASYNC
  v4u aR0, aR1;        // A tile register double buffer (sync fallback)
#endif

  // ---- prologue: stage tile 0 into buffer 0 ----
  {
    const v4u* gA = reinterpret_cast<const v4u*>(A + (size_t)(bm + a_row) * lda);
#if HAVE_ASYNC
    async_cp16(gA + a_pg,     &sA[0][a_row * SA_STRIDE + a_pg * 4]);
    async_cp16(gA + a_pg + 1, &sA[0][a_row * SA_STRIDE + a_pg * 4 + 4]);
#else
    aR0 = gA[a_pg]; aR1 = gA[a_pg + 1];
    *reinterpret_cast<v4u*>(&sA[0][a_row * SA_STRIDE + a_pg * 4])     = aR0;
    *reinterpret_cast<v4u*>(&sA[0][a_row * SA_STRIDE + a_pg * 4 + 4]) = aR1;
#endif
    bR0 = loadB_unit(0, b_p0, b_c0);
    bR1 = loadB_unit(0, b_p1, b_c1);
    *reinterpret_cast<v4u*>(&sB[0][b_p0 * SB_STRIDE + b_c0]) = bR0;
    *reinterpret_cast<v4u*>(&sB[0][b_p1 * SB_STRIDE + b_c1]) = bR1;
  }
  async_wait0();
  __syncthreads();

  for (int t = 0; t < nk; ++t) {
    const int cur = t & 1;
    const int nxt = cur ^ 1;
    const bool more = (t + 1 < nk);
    const int kkn = (t + 1) * GBK;

    // ---- issue next tile while current computes ----
    if (more) {
      const v4u* gA = reinterpret_cast<const v4u*>(
          A + (size_t)(bm + a_row) * lda + kkn);
#if HAVE_ASYNC
      async_cp16(gA + a_pg,     &sA[nxt][a_row * SA_STRIDE + a_pg * 4]);
      async_cp16(gA + a_pg + 1, &sA[nxt][a_row * SA_STRIDE + a_pg * 4 + 4]);
#else
      aR0 = gA[a_pg]; aR1 = gA[a_pg + 1];
#endif
      bR0 = loadB_unit(kkn, b_p0, b_c0);
      bR1 = loadB_unit(kkn, b_p1, b_c1);
      if (t + 2 < nk)
        __builtin_prefetch(
            (const void*)(A + (size_t)(bm + a_row) * lda + kkn + GBK), 0, 2);
    }

    // ---- compute from current buffer ----
    v8u af[2];
#pragma unroll
    for (int mt = 0; mt < 2; ++mt) {
      int row = wr * 32 + mt * 16 + l16;
      int p0  = half * 8;
      v4u lo = *reinterpret_cast<const v4u*>(&sA[cur][row * SA_STRIDE + p0]);
      v4u hi = *reinterpret_cast<const v4u*>(&sA[cur][row * SA_STRIDE + p0 + 4]);
      af[mt] = __builtin_shufflevector(lo, hi, 0, 1, 2, 3, 4, 5, 6, 7);
    }
#pragma unroll
    for (int nt = 0; nt < 4; ++nt) {
      int col = wc * 64 + nt * 16 + l16;
      int pb  = half * 8;
      v8u bf;
#pragma unroll
      for (int j = 0; j < 8; ++j) bf[j] = sB[cur][(pb + j) * SB_STRIDE + col];
#pragma unroll
      for (int mt = 0; mt < 2; ++mt) acc[mt][nt] = wmma_bf16(af[mt], bf, acc[mt][nt]);
    }

    // ---- commit next tile ----
    if (more) {
#if !HAVE_ASYNC
      *reinterpret_cast<v4u*>(&sA[nxt][a_row * SA_STRIDE + a_pg * 4])     = aR0;
      *reinterpret_cast<v4u*>(&sA[nxt][a_row * SA_STRIDE + a_pg * 4 + 4]) = aR1;
#endif
      *reinterpret_cast<v4u*>(&sB[nxt][b_p0 * SB_STRIDE + b_c0]) = bR0;
      *reinterpret_cast<v4u*>(&sB[nxt][b_p1 * SB_STRIDE + b_c1]) = bR1;
      async_wait0();
    }
    __syncthreads();
  }

  // ---- epilogue: C layout VGPR r <-> row r + 8*half, col = lane%16 ----
#pragma unroll
  for (int mt = 0; mt < 2; ++mt)
#pragma unroll
    for (int nt = 0; nt < 4; ++nt) {
      int row0 = bm + wr * 32 + mt * 16 + half * 8;
      int col  = bn + wc * 64 + nt * 16 + l16;
      float bv = bias ? bias[col] : 0.0f;
#pragma unroll
      for (int r = 0; r < 8; ++r) {
        float vv = acc[mt][nt][r] + bv;
        size_t idx = (size_t)(row0 + r) * N + col;
        if (outF) outF[idx] = vv;
        if (outH) outH[idx] = f2bf(vv);
      }
    }
}

// ---------------------------------------------------------------------------
// Flash attention: one block per (b, h, 128 query rows). d=64, T=1024.
// logits = (q.k)/d (reference applies 1/sqrt(d) twice), causal mask, online
// softmax, P@V in bf16 WMMA, f32 accumulators.
// ---------------------------------------------------------------------------
#define QROWS 128
#define KT 64
#define SQ_STRIDE 36   // 32 d-pairs + pad
#define SK_STRIDE 36
#define SV_STRIDE 68   // 64 d-cols + pad
#define SP_STRIDE 36   // 32 key-pairs + pad

__global__ __launch_bounds__(256)
void k_attn(const u16* __restrict__ q, const u16* __restrict__ k,
            const u16* __restrict__ v, float* __restrict__ o,
            int T, int D, int Hn, int Cn) {
  __shared__ u32 sQ[QROWS * SQ_STRIDE];
  __shared__ u32 sK[KT * SK_STRIDE];
  __shared__ u32 sV[32 * SV_STRIDE];        // key-pair interleaved
  __shared__ u32 sP[8 * 16 * SP_STRIDE];    // per-wave P staging

  const int tid  = threadIdx.x;
  const int lane = tid & 31;
  const int wave = tid >> 5;
  const int half = lane >> 4;
  const int l16  = lane & 15;
  const int qs   = blockIdx.x * QROWS;
  const int b    = blockIdx.y >> 4;
  const int h    = blockIdx.y & 15;

  const u16* qb = q + (size_t)(b * T + qs) * Cn + h * D;
  const u16* kb = k + (size_t)(b * T) * Cn + h * D;
  const u16* vb = v + (size_t)(b * T) * Cn + h * D;

  // ---- stage Q block (128 rows x 32 u32 pairs) ----
#pragma unroll
  for (int uu = 0; uu < 4; ++uu) {
    int u   = tid + uu * 256;       // 0..1023
    int row = u >> 3;
    int cg  = u & 7;
#if HAVE_ASYNC
    async_cp16(reinterpret_cast<const v4u*>(qb + (size_t)row * Cn) + cg,
               &sQ[row * SQ_STRIDE + cg * 4]);
#else
    v4u x = reinterpret_cast<const v4u*>(qb + (size_t)row * Cn)[cg];
    *reinterpret_cast<v4u*>(&sQ[row * SQ_STRIDE + cg * 4]) = x;
#endif
  }
  async_wait0();
  __syncthreads();

  // ---- Q A-fragments held in registers for whole key loop ----
  v8u aq[2];
  {
    int row = wave * 16 + l16;
#pragma unroll
    for (int ds = 0; ds < 2; ++ds) {
      int p0 = ds * 16 + half * 8;
      v4u lo = *reinterpret_cast<const v4u*>(&sQ[row * SQ_STRIDE + p0]);
      v4u hi = *reinterpret_cast<const v4u*>(&sQ[row * SQ_STRIDE + p0 + 4]);
      aq[ds] = __builtin_shufflevector(lo, hi, 0, 1, 2, 3, 4, 5, 6, 7);
    }
  }

  float mrow[8], lrow[8];
  v8f oacc[4];
#pragma unroll
  for (int r = 0; r < 8; ++r) { mrow[r] = -1e30f; lrow[r] = 0.0f; }
#pragma unroll
  for (int nt = 0; nt < 4; ++nt) oacc[nt] = (v8f){0, 0, 0, 0, 0, 0, 0, 0};

  const float inv_dd = 1.0f / 64.0f;           // (1/sqrt(d))^2, d=64
  const int ntiles = qs / KT + 2;              // causal key-tile bound

  for (int jt = 0; jt < ntiles; ++jt) {
    const int j0 = jt * KT;
    __syncthreads();
    // ---- stage K tile row-major (its layout IS the QK^T B-fragment) ----
#pragma unroll
    for (int uu = 0; uu < 2; ++uu) {
      int u = tid + uu * 256;       // 0..511
      int row = u >> 3;
      int cg  = u & 7;
#if HAVE_ASYNC
      async_cp16(reinterpret_cast<const v4u*>(kb + (size_t)(j0 + row) * Cn) + cg,
                 &sK[row * SK_STRIDE + cg * 4]);
#else
      v4u x = reinterpret_cast<const v4u*>(kb + (size_t)(j0 + row) * Cn)[cg];
      *reinterpret_cast<v4u*>(&sK[row * SK_STRIDE + cg * 4]) = x;
#endif
    }
    // ---- stage V tile, key-pair interleaved ----
#pragma unroll
    for (int uu = 0; uu < 2; ++uu) {
      int u   = tid + uu * 256;
      int p   = u >> 4;             // key pair 0..31
      int col = (u & 15) * 4;       // 0..60
      const u16* g0 = vb + (size_t)(j0 + 2 * p) * Cn + col;
      uint2 r0 = *reinterpret_cast<const uint2*>(g0);
      uint2 r1 = *reinterpret_cast<const uint2*>(g0 + Cn);
      v4u w;
      w.x = (r0.x & 0xFFFFu) | (r1.x << 16);
      w.y = (r0.x >> 16)     | (r1.x & 0xFFFF0000u);
      w.z = (r0.y & 0xFFFFu) | (r1.y << 16);
      w.w = (r0.y >> 16)     | (r1.y & 0xFFFF0000u);
      *reinterpret_cast<v4u*>(&sV[p * SV_STRIDE + col]) = w;
    }
    async_wait0();
    __syncthreads();

    // ---- S = Q @ K^T (16 rows x 64 keys per wave) ----
    v8f s[4];
#pragma unroll
    for (int nt = 0; nt < 4; ++nt) s[nt] = (v8f){0, 0, 0, 0, 0, 0, 0, 0};
#pragma unroll
    for (int ds = 0; ds < 2; ++ds) {
#pragma unroll
      for (int nt = 0; nt < 4; ++nt) {
        int krow = nt * 16 + l16;
        int pb   = ds * 16 + half * 8;
        v8u bk;
#pragma unroll
        for (int j = 0; j < 8; ++j) bk[j] = sK[krow * SK_STRIDE + pb + j];
        s[nt] = wmma_bf16(aq[ds], bk, s[nt]);
      }
    }

    // ---- online softmax ----
    float pl[4][8], mloc[8];
#pragma unroll
    for (int r = 0; r < 8; ++r) mloc[r] = -1e30f;
#pragma unroll
    for (int nt = 0; nt < 4; ++nt)
#pragma unroll
      for (int r = 0; r < 8; ++r) {
        float x = s[nt][r] * inv_dd;
        int qg = qs + wave * 16 + r + 8 * half;
        int kg = j0 + nt * 16 + l16;
        if (kg > qg) x = -1e30f;        // causal mask
        pl[nt][r] = x;
        mloc[r] = fmaxf(mloc[r], x);
      }
#pragma unroll
    for (int r = 0; r < 8; ++r) {
      float mn = fmaxf(mrow[r], hmax16(mloc[r]));
      float alpha = __expf(mrow[r] - mn);
      mrow[r] = mn;
      lrow[r] *= alpha;
#pragma unroll
      for (int nt = 0; nt < 4; ++nt) oacc[nt][r] *= alpha;
    }
    float rs[8];
#pragma unroll
    for (int r = 0; r < 8; ++r) rs[r] = 0.0f;
#pragma unroll
    for (int nt = 0; nt < 4; ++nt)
#pragma unroll
      for (int r = 0; r < 8; ++r) {
        float p = __expf(pl[nt][r] - mrow[r]);
        rs[r] += p;
        int prow = r + 8 * half;                       // C-layout -> row-major P
        reinterpret_cast<u16*>(&sP[(wave * 16 + prow) * SP_STRIDE])[nt * 16 + l16] =
            f2bf(p);
      }
#pragma unroll
    for (int r = 0; r < 8; ++r) lrow[r] += hsum16(rs[r]);

    // ---- O += P @ V (per-wave P staged through LDS: C-layout -> A-layout) ----
#pragma unroll
    for (int kst = 0; kst < 2; ++kst) {
      int prow = wave * 16 + l16;
      int p0   = kst * 16 + half * 8;
      v4u lo = *reinterpret_cast<const v4u*>(&sP[prow * SP_STRIDE + p0]);
      v4u hi = *reinterpret_cast<const v4u*>(&sP[prow * SP_STRIDE + p0 + 4]);
      v8u pa = __builtin_shufflevector(lo, hi, 0, 1, 2, 3, 4, 5, 6, 7);
#pragma unroll
      for (int nt = 0; nt < 4; ++nt) {
        int col = nt * 16 + l16;
        int pb  = kst * 16 + half * 8;
        v8u bv;
#pragma unroll
        for (int j = 0; j < 8; ++j) bv[j] = sV[(pb + j) * SV_STRIDE + col];
        oacc[nt] = wmma_bf16(pa, bv, oacc[nt]);
      }
    }
  }

  // ---- write O (layout b,n,h,d == rows of [B*T, C] with head col offset) ----
#pragma unroll
  for (int nt = 0; nt < 4; ++nt)
#pragma unroll
    for (int r = 0; r < 8; ++r) {
      int qg  = qs + wave * 16 + r + 8 * half;
      int col = h * D + nt * 16 + l16;
      o[(size_t)(b * T + qg) * Cn + col] = oacc[nt][r] / lrow[r];
    }
}

// ---------------------------------------------------------------------------
// Row LayerNorm (eps 1e-5) -> bf16 for the final WMMA GEMM
// ---------------------------------------------------------------------------
__global__ __launch_bounds__(256)
void k_layernorm_bf16(const float* __restrict__ x, const float* __restrict__ g,
                      const float* __restrict__ bta, u16* __restrict__ y, int C) {
  const int row = blockIdx.x;
  const int tid = threadIdx.x;
  const float* xr = x + (size_t)row * C;
  float s = 0.0f, s2 = 0.0f;
  for (int i = tid; i < C; i += 256) {
    float vv = xr[i];
    s += vv;
    s2 += vv * vv;
  }
  s  = wsum32(s);
  s2 = wsum32(s2);
  __shared__ float red[16];
  if ((tid & 31) == 0) { red[tid >> 5] = s; red[8 + (tid >> 5)] = s2; }
  __syncthreads();
  float S = 0.0f, S2 = 0.0f;
#pragma unroll
  for (int i = 0; i < 8; ++i) { S += red[i]; S2 += red[8 + i]; }
  float mu   = S / C;
  float var  = S2 / C - mu * mu;
  float rstd = rsqrtf(var + 1e-5f);
  for (int i = tid; i < C; i += 256) {
    float vv = (xr[i] - mu) * rstd * g[i] + bta[i];
    y[(size_t)row * C + i] = f2bf(vv);
  }
}

// ---------------------------------------------------------------------------
// Host orchestration
// ---------------------------------------------------------------------------
extern "C" void kernel_launch(void* const* d_in, const int* in_sizes, int n_in,
                              void* d_out, int out_size, void* d_ws, size_t ws_size,
                              hipStream_t stream) {
  (void)in_sizes; (void)n_in; (void)out_size; (void)ws_size;
  const int B = 4, T = 1024, C = 1024, Hh = 16, D = 64;
  const int M = B * T;                       // 4096

  const float* x        = (const float*)d_in[0];
  const float* c_attn_w = (const float*)d_in[1];
  const float* c_attn_b = (const float*)d_in[2];
  const float* q_w      = (const float*)d_in[3];
  const float* q_b      = (const float*)d_in[4];
  const float* k_w      = (const float*)d_in[5];
  const float* k_b      = (const float*)d_in[6];
  const float* v_w      = (const float*)d_in[7];
  const float* v_b      = (const float*)d_in[8];
  const float* ln_g     = (const float*)d_in[9];
  const float* ln_b     = (const float*)d_in[10];
  const float* out_w    = (const float*)d_in[11];
  const float* out_b    = (const float*)d_in[12];

  char* ws = (char*)d_ws;
  size_t off = 0;
  auto alloc = [&](size_t bytes) { char* p = ws + off; off += (bytes + 255) & ~size_t(255); return p; };
  u16*   x16    = (u16*)alloc((size_t)M * C * 2);
  u16*   wat16  = (u16*)alloc((size_t)C * 3 * C * 2);
  u16*   qw16   = (u16*)alloc((size_t)C * C * 2);
  u16*   kw16   = (u16*)alloc((size_t)C * C * 2);
  u16*   vw16   = (u16*)alloc((size_t)C * C * 2);
  u16*   ow16   = (u16*)alloc((size_t)C * C * 2);
  u16*   qkv16  = (u16*)alloc((size_t)M * 3 * C * 2);
  u16*   q16    = (u16*)alloc((size_t)M * C * 2);
  u16*   k16    = (u16*)alloc((size_t)M * C * 2);
  u16*   v16    = (u16*)alloc((size_t)M * C * 2);
  float* o32    = (float*)alloc((size_t)M * C * 4);
  u16*   ln16   = (u16*)alloc((size_t)M * C * 2);

  auto cvt = [&](const float* src, u16* dst, int n) {
    int n4 = n / 4;
    int blocks = (n4 + 255) / 256;
    if (blocks > 2048) blocks = 2048;
    k_f32_to_bf16<<<blocks, 256, 0, stream>>>(src, dst, n4);
  };
  cvt(x, x16, M * C);
  cvt(c_attn_w, wat16, C * 3 * C);
  cvt(q_w, qw16, C * C);
  cvt(k_w, kw16, C * C);
  cvt(v_w, vw16, C * C);
  cvt(out_w, ow16, C * C);

  // qkv = x @ c_attn_w + c_attn_b  -> bf16 [4096, 3072]
  k_gemm_bf16<<<dim3(3 * C / GBN, M / GBM), 256, 0, stream>>>(
      x16, C, wat16, 3 * C, c_attn_b, nullptr, qkv16, M, 3 * C, C);
  // q/k/v projections from qkv stream slices (lda = 3C)
  k_gemm_bf16<<<dim3(C / GBN, M / GBM), 256, 0, stream>>>(
      qkv16 + 0 * C, 3 * C, qw16, C, q_b, nullptr, q16, M, C, C);
  k_gemm_bf16<<<dim3(C / GBN, M / GBM), 256, 0, stream>>>(
      qkv16 + 1 * C, 3 * C, kw16, C, k_b, nullptr, k16, M, C, C);
  k_gemm_bf16<<<dim3(C / GBN, M / GBM), 256, 0, stream>>>(
      qkv16 + 2 * C, 3 * C, vw16, C, v_b, nullptr, v16, M, C, C);

  // causal attention -> o32 [4096, 1024] (b,n,h,d layout)
  k_attn<<<dim3(T / QROWS, B * Hh), 256, 0, stream>>>(q16, k16, v16, o32, T, D, Hh, C);

  // LayerNorm -> bf16
  k_layernorm_bf16<<<M, 256, 0, stream>>>(o32, ln_g, ln_b, ln16, C);

  // out = ln @ out_w + out_b -> f32 d_out
  k_gemm_bf16<<<dim3(C / GBN, M / GBM), 256, 0, stream>>>(
      ln16, C, ow16, C, out_b, (float*)d_out, nullptr, M, C, C);
}
